// MPNNConv_48971217109453
// MI455X (gfx1250) — matrix operand order repaired and checked by reading, not compiled, not run on hardware
//
#include <hip/hip_runtime.h>
#include <stdint.h>

#define D_DIM 256

typedef __attribute__((ext_vector_type(8)))  float   v8f;
typedef __attribute__((ext_vector_type(16))) __bf16  v16bf;

// ---------- bf16 helpers (round-to-nearest-even) ----------
__device__ __forceinline__ uint16_t f2bf(float f) {
  uint32_t u = __float_as_uint(f);
  u = u + 0x7FFFu + ((u >> 16) & 1u);
  return (uint16_t)(u >> 16);
}
__device__ __forceinline__ uint32_t pack_bf16x2(float a, float b) {
  uint32_t ua = __float_as_uint(a);
  uint32_t ub = __float_as_uint(b);
  ua = ua + 0x7FFFu + ((ua >> 16) & 1u);
  ub = ub + 0x7FFFu + ((ub >> 16) & 1u);
  return (ua >> 16) | (ub & 0xFFFF0000u);
}

// LDS byte offset of a pointer to __shared__ (aperture tag is in the high
// dword of the generic address; low 32 bits are the LDS offset).
__device__ __forceinline__ uint32_t lds_off(const void* p) {
  return (uint32_t)(uintptr_t)p;
}

// ---------- fp32 -> bf16 weight conversion ----------
__global__ void cvt_bf16_kernel(const float* __restrict__ src,
                                uint16_t* __restrict__ dst, int n) {
  int i = blockIdx.x * blockDim.x + threadIdx.x;
  if (i < n) dst[i] = f2bf(src[i]);
}

// ---------- zero buffer ----------
__global__ void zero_kernel(float4* __restrict__ p, int n4) {
  int i = blockIdx.x * blockDim.x + threadIdx.x;
  if (i < n4) p[i] = make_float4(0.f, 0.f, 0.f, 0.f);
}

// ---------- Tiled WMMA GEMM:  Out = act(X @ W^T + bias) ----------
// X: [rows x K] fp32 (row-major), W: bf16 packed pairs [M x K/2] (row-major),
// Out: [rows x M] fp32. Block = 256 threads (8 wave32s), tile 128x64.
// Each wave: 16 rows x 64 cols = four 16x16 f32 accumulators.
// B-tile staged memory->LDS with GLOBAL_LOAD_ASYNC_TO_LDS_B32 (ASYNCcnt);
// A-tile staged through VALU (fp32->bf16 conversion) with prefetch of the
// next K-chunk.
__global__ __launch_bounds__(256)
void gemm_bf16_wmma_kernel(const float* __restrict__ X,
                           const uint32_t* __restrict__ Wp,
                           const float* __restrict__ bias,
                           float* __restrict__ Out,
                           int rows, int K, int M, int relu) {
  __shared__ uint32_t As[128 * 17];  // 128 rows x 16 bf16-pairs (+1 pad)
  __shared__ uint32_t Bs[64 * 17];   // 64 cols  x 16 bf16-pairs (+1 pad)

  const int tid    = threadIdx.x;
  const int lane   = tid & 31;       // wave32
  const int wave   = tid >> 5;
  const int rowBlk = blockIdx.x * 128;
  const int colBlk = blockIdx.y * 64;
  const int Kp     = K >> 1;

  v8f acc[4] = {};

  const bool hi  = lane >= 16;
  const int  l15 = lane & 15;

  for (int kk = 0; kk < K; kk += 32) {
    // ---- async stage B tile: 64x16 words, 4 per thread, no VGPR round-trip
#pragma unroll
    for (int t = 0; t < 4; ++t) {
      const int i = tid + t * 256;          // 0..1023, all lanes active
      const int c = i >> 4, p = i & 15;
      const uint32_t ldsb = lds_off(&Bs[c * 17 + p]);
      const uint64_t gaddr =
          (uint64_t)(uintptr_t)(Wp + (size_t)(colBlk + c) * Kp + (kk >> 1) + p);
      asm volatile("global_load_async_to_lds_b32 %0, %1, off"
                   :: "v"(ldsb), "v"(gaddr) : "memory");
    }
    // ---- stage A tile (fp32 -> packed bf16 pairs) ----
    for (int i = tid; i < 128 * 16; i += 256) {
      int r = i >> 4, p = i & 15;
      int grow = rowBlk + r;
      uint32_t w = 0u;
      if (grow < rows) {
        const float* xp = X + (size_t)grow * K + kk + 2 * p;
        const float2 v = *(const float2*)xp;
        w = pack_bf16x2(v.x, v.y);
        if (kk + 32 < K) __builtin_prefetch(xp + 32, 0, 1);  // next K-chunk
      }
      As[r * 17 + p] = w;
    }
    asm volatile("s_wait_asynccnt 0x0" ::: "memory");
    __syncthreads();

    // ---- build A fragment (16x32 bf16, contiguous k = 16h + s) ----
    union { v16bf v; uint32_t u[8]; } a;
    const int arow = wave * 16 + l15;
    const int ap0  = hi ? 8 : 0;     // lanes 16-31 hold K 16..31
#pragma unroll
    for (int i = 0; i < 4; ++i) {
      a.u[i]     = As[arow * 17 + ap0 + i];
      a.u[4 + i] = As[arow * 17 + ap0 + 4 + i];
    }

    // ---- four B fragments (32x16 each) + WMMA ----
    const int bp0 = hi ? 8 : 0;      // lanes 16-31 hold K 16..31
#pragma unroll
    for (int nb = 0; nb < 4; ++nb) {
      union { v16bf v; uint32_t u[8]; } b;
      const int bcol = nb * 16 + l15;
#pragma unroll
      for (int i = 0; i < 8; ++i) b.u[i] = Bs[bcol * 17 + bp0 + i];
      acc[nb] = __builtin_amdgcn_wmma_f32_16x16x32_bf16(
          false, a.v, false, b.v, (short)0, acc[nb], false, false);
    }
    __syncthreads();
  }

  // ---- epilogue: bias (+ ReLU), C/D layout: lane = N + 16*(M>=8) ----
  const int rbase = rowBlk + wave * 16 + (hi ? 8 : 0);
#pragma unroll
  for (int nb = 0; nb < 4; ++nb) {
    const int col = colBlk + nb * 16 + l15;
    const float bs = bias[col];
#pragma unroll
    for (int r = 0; r < 8; ++r) {
      const int row = rbase + r;
      if (row < rows) {
        float v = acc[nb][r] + bs;
        if (relu) v = fmaxf(v, 0.0f);
        Out[(size_t)row * M + col] = v;
      }
    }
  }
}

// ---------- edge scatter: msg[dst] += h[src]  (wave per edge) ----------
__global__ __launch_bounds__(256)
void edge_scatter_kernel(const float* __restrict__ h,
                         const int* __restrict__ src,
                         const int* __restrict__ dst,
                         float* __restrict__ msg, int E) {
  const int e    = blockIdx.x * 8 + (threadIdx.x >> 5);
  const int lane = threadIdx.x & 31;
  if (e >= E) return;
  const int s = src[e], d = dst[e];
  const float* hp = h   + (size_t)s * D_DIM;
  float*       mp = msg + (size_t)d * D_DIM;
#pragma unroll
  for (int j = 0; j < D_DIM / 32; ++j) {
    const int idx = lane + j * 32;           // coalesced within wave
    atomicAdd(&mp[idx], hp[idx]);            // global_atomic_add_f32 (L2)
  }
}

// ---------- fused GRU elementwise update ----------
__global__ __launch_bounds__(256)
void gru_kernel(const float* __restrict__ gx, const float* __restrict__ gh,
                const float* __restrict__ X, float* __restrict__ out, int n) {
  const int i = blockIdx.x * blockDim.x + threadIdx.x;
  if (i >= n) return;
  const int row = i >> 8;            // D = 256
  const int j   = i & 255;
  const size_t base = (size_t)row * (3 * D_DIM) + j;
  const float xr = gx[base], xz = gx[base + D_DIM], xn = gx[base + 2 * D_DIM];
  const float hr = gh[base], hz = gh[base + D_DIM], hn = gh[base + 2 * D_DIM];
  const float r  = 1.0f / (1.0f + __expf(-(xr + hr)));
  const float z  = 1.0f / (1.0f + __expf(-(xz + hz)));
  const float nv = tanhf(xn + r * hn);
  out[i] = (1.0f - z) * nv + z * X[i];
}

extern "C" void kernel_launch(void* const* d_in, const int* in_sizes, int n_in,
                              void* d_out, int out_size, void* d_ws, size_t ws_size,
                              hipStream_t stream) {
  (void)n_in; (void)out_size; (void)ws_size;
  const int D  = D_DIM;
  const int N  = in_sizes[0] / D;
  const int E  = in_sizes[9];

  const float* node_feats = (const float*)d_in[0];
  const float* W1   = (const float*)d_in[1];
  const float* b1   = (const float*)d_in[2];
  const float* W2   = (const float*)d_in[3];
  const float* b2   = (const float*)d_in[4];
  const float* W_ih = (const float*)d_in[5];
  const float* b_ih = (const float*)d_in[6];
  const float* W_hh = (const float*)d_in[7];
  const float* b_hh = (const float*)d_in[8];
  const int*   src  = (const int*)d_in[9];
  const int*   dst  = (const int*)d_in[10];
  float* out = (float*)d_out;

  // workspace layout (floats)
  const size_t ND = (size_t)N * D;
  float* ws   = (float*)d_ws;
  float* Xbuf = ws;                // N*D
  float* hbuf = ws + ND;           // N*D
  float* msg  = ws + 2 * ND;       // N*D
  float* gx   = ws + 3 * ND;       // 3*N*D (first N*D doubles as h1 staging)
  float* gh   = ws + 6 * ND;       // 3*N*D
  uint16_t* wq   = (uint16_t*)(ws + 9 * ND);
  uint16_t* W1b  = wq;
  uint16_t* W2b  = W1b + (size_t)D * D;
  uint16_t* Wihb = W2b + (size_t)D * D;
  uint16_t* Whhb = Wihb + (size_t)3 * D * D;

  const dim3 blk(256);
  // weights -> bf16 (once per launch; deterministic)
  cvt_bf16_kernel<<<(D * D + 255) / 256, blk, 0, stream>>>(W1, W1b, D * D);
  cvt_bf16_kernel<<<(D * D + 255) / 256, blk, 0, stream>>>(W2, W2b, D * D);
  cvt_bf16_kernel<<<(3 * D * D + 255) / 256, blk, 0, stream>>>(W_ih, Wihb, 3 * D * D);
  cvt_bf16_kernel<<<(3 * D * D + 255) / 256, blk, 0, stream>>>(W_hh, Whhb, 3 * D * D);

  hipMemcpyAsync(Xbuf, node_feats, ND * sizeof(float),
                 hipMemcpyDeviceToDevice, stream);

  const dim3 g256((N + 127) / 128, D / 64);        // M = 256
  const dim3 g768((N + 127) / 128, (3 * D) / 64);  // M = 768

  for (int step = 0; step < 6; ++step) {
    float* h1 = gx;  // reuse gx region for the MLP hidden layer
    // h1 = relu(X @ W1^T + b1)
    gemm_bf16_wmma_kernel<<<g256, blk, 0, stream>>>(
        Xbuf, (const uint32_t*)W1b, b1, h1, N, D, D, 1);
    // h = h1 @ W2^T + b2
    gemm_bf16_wmma_kernel<<<g256, blk, 0, stream>>>(
        h1, (const uint32_t*)W2b, b2, hbuf, N, D, D, 0);
    // msg = segment_sum(h[src], dst)
    zero_kernel<<<((int)(ND / 4) + 255) / 256, blk, 0, stream>>>(
        (float4*)msg, (int)(ND / 4));
    edge_scatter_kernel<<<(E + 7) / 8, blk, 0, stream>>>(hbuf, src, dst, msg, E);
    // gx = msg @ W_ih^T + b_ih ; gh = X @ W_hh^T + b_hh
    gemm_bf16_wmma_kernel<<<g768, blk, 0, stream>>>(
        msg, (const uint32_t*)Wihb, b_ih, gx, N, D, 3 * D, 0);
    gemm_bf16_wmma_kernel<<<g768, blk, 0, stream>>>(
        Xbuf, (const uint32_t*)Whhb, b_hh, gh, N, D, 3 * D, 0);
    // GRU update (in-place on Xbuf; final step writes d_out)
    float* tgt = (step == 5) ? out : Xbuf;
    gru_kernel<<<((int)ND + 255) / 256, blk, 0, stream>>>(gx, gh, Xbuf, tgt, (int)ND);
  }
}
